// Head_3049426780247
// MI455X (gfx1250) — matrix operand order repaired and checked by reading, not compiled, not run on hardware
//
#include <hip/hip_runtime.h>
#include <math.h>

// ---- CDNA5 WMMA types -------------------------------------------------------
typedef __bf16 v16bf __attribute__((ext_vector_type(16)));
typedef float  v8f   __attribute__((ext_vector_type(8)));

#define HEAD 64
#define EMBD 128
#define BB   16
#define TT   2048
#define WPB  8          // waves per block (256 threads, wave32)

// fp32 -> bf16 via hardware convert (v_cvt), not manual bit twiddling.
__device__ __forceinline__ unsigned short f2bf(float f) {
  union { __bf16 h; unsigned short s; } c;
  c.h = (__bf16)f;
  return c.s;
}

__device__ __forceinline__ v16bf pack16(const unsigned short* s) {
  union { v16bf v; unsigned short s[16]; } u;
#pragma unroll
  for (int j = 0; j < 16; ++j) u.s[j] = s[j];
  return u.v;
}

// A-matrix 16x32 bf16 fragment (this file's contiguous k order): lane L holds row M=L&15,
// lanes 0-15: K=0..15, lanes 16-31: K=16..31 (matches B fragments).
__device__ __forceinline__ v16bf ldsAfrag(const unsigned short* t, int ld, int col0, int lane) {
  int m  = lane & 15;
  int kb = (lane < 16) ? 0 : 16;
  unsigned short tmp[16];
#pragma unroll
  for (int j = 0; j < 16; ++j) tmp[j] = t[m * ld + col0 + kb + j];
  return pack16(tmp);
}

// B-matrix 32x16 bf16 fragment: lane L holds column N=L&15,
// lanes 0-15: K=0..15, lanes 16-31: K=16..31.  B(k,n) = t[(row0+k)*ld + col0+n]
__device__ __forceinline__ v16bf ldsBfrag(const unsigned short* t, int ld, int row0, int col0, int lane) {
  int n  = lane & 15;
  int kb = (lane < 16) ? 0 : 16;
  unsigned short tmp[16];
#pragma unroll
  for (int j = 0; j < 16; ++j) tmp[j] = t[(row0 + kb + j) * ld + col0 + n];
  return pack16(tmp);
}

// B fragment of K^T: B(h, s) = Ktile[s, h]; row0 = key offset, col0 = h offset.
__device__ __forceinline__ v16bf ldsBTfrag(const unsigned short* t, int ld, int row0, int col0, int lane) {
  int n  = lane & 15;             // key column of S tile
  int kb = (lane < 16) ? 0 : 16;  // h within 32-chunk
  unsigned short tmp[16];
#pragma unroll
  for (int j = 0; j < 16; ++j) tmp[j] = t[(row0 + n) * ld + col0 + kb + j];
  return pack16(tmp);
}

__device__ __forceinline__ v8f wmma_bf16(v16bf a, v16bf b, v8f c) {
  return __builtin_amdgcn_wmma_f32_16x16x32_bf16(false, a, false, b, (short)0, c, false, false);
}

__device__ __forceinline__ v8f zero8() {
  v8f z;
#pragma unroll
  for (int i = 0; i < 8; ++i) z[i] = 0.0f;
  return z;
}

// ---- Kernel 0: pre-pack Wq/Wk/Wv into B-fragment order (bf16) --------------
// 48 fragments: w(3) x kchunk(4) x ntile(4); each = 32 lanes x 16 bf16, so the
// projection kernel reads weights as contiguous coalesced 32 B/lane fragments.
__global__ void prep_wfrag(const float* __restrict__ Wq, const float* __restrict__ Wk,
                           const float* __restrict__ Wv, unsigned short* __restrict__ wfrag) {
  int f = blockIdx.x;              // 0..47
  int lane = threadIdx.x;          // 0..31
  int w  = f >> 4;                 // 0=q,1=k,2=v
  int kc = (f >> 2) & 3;
  int nt = f & 3;
  const float* W = (w == 0) ? Wq : (w == 1) ? Wk : Wv;
  int n  = lane & 15;
  int kb = (lane < 16) ? 0 : 16;
  int col = nt * 16 + n;
#pragma unroll
  for (int j = 0; j < 16; ++j) {
    int c = kc * 32 + kb + j;
    wfrag[f * 512 + lane * 16 + j] = f2bf(W[c * HEAD + col]);
  }
}

// ---- Kernel 1: fused QKV projection (x[16x128] @ W[128x64] + b) ------------
// One wave per 16-row tile; 8 waves per block. 1/sqrt(H) folded into Q.
__global__ __launch_bounds__(256) void proj_qkv(
    const float* __restrict__ x,
    const float* __restrict__ bq, const float* __restrict__ bk, const float* __restrict__ bv,
    const unsigned short* __restrict__ wfrag,
    unsigned short* __restrict__ qb, unsigned short* __restrict__ kbuf,
    unsigned short* __restrict__ vbuf) {
  __shared__ unsigned short xs[WPB][16 * EMBD];   // 32 KB
  int lane = threadIdx.x & 31;
  int wid  = threadIdx.x >> 5;
  int tile = blockIdx.x * WPB + wid;              // 0..2047
  int b     = tile >> 7;
  int qBase = (tile & 127) * 16;

  const float* xg = x + ((size_t)(b * TT + qBase)) * EMBD;
  unsigned short* xw = xs[wid];
  // Stage x tile (16x128) as bf16 into LDS, 16B-vectorized loads.
#pragma unroll
  for (int i = 0; i < 16; ++i) {
    float4 v = ((const float4*)xg)[i * 32 + lane];
    ushort4 u;
    u.x = f2bf(v.x); u.y = f2bf(v.y); u.z = f2bf(v.z); u.w = f2bf(v.w);
    ((ushort4*)xw)[i * 32 + lane] = u;
  }
  asm volatile("s_wait_dscnt 0" ::: "memory");

  v16bf a[4];
#pragma unroll
  for (int kc = 0; kc < 4; ++kc) a[kc] = ldsAfrag(xw, EMBD, kc * 32, lane);

  int n   = lane & 15;
  int hi8 = (lane < 16) ? 0 : 8;
  for (int w = 0; w < 3; ++w) {
    v8f acc[4];
#pragma unroll
    for (int nt = 0; nt < 4; ++nt) acc[nt] = zero8();
#pragma unroll
    for (int kc = 0; kc < 4; ++kc) {
#pragma unroll
      for (int nt = 0; nt < 4; ++nt) {
        // Weight fragments are pre-packed: contiguous 32B per lane.
        v16bf bf = pack16(wfrag + (((w * 4 + kc) * 4 + nt) * 512) + lane * 16);
        acc[nt] = wmma_bf16(a[kc], bf, acc[nt]);
      }
    }
    const float* bias = (w == 0) ? bq : (w == 1) ? bk : bv;
    unsigned short* obuf = (w == 0) ? qb : (w == 1) ? kbuf : vbuf;
    float s = (w == 0) ? 0.125f : 1.0f;   // fold 1/sqrt(64) into Q
#pragma unroll
    for (int nt = 0; nt < 4; ++nt) {
      int col = nt * 16 + n;
      float bc = bias[col];
#pragma unroll
      for (int r = 0; r < 8; ++r) {
        int m = r + hi8;                  // f32 C-layout: row = r (+8 for hi half)
        obuf[((size_t)(b * TT + qBase + m)) * HEAD + col] = f2bf((acc[nt][r] + bc) * s);
      }
    }
  }
}

// ---- Kernel 2: flash attention, one wave per 16-query tile -----------------
// Separate K and V LDS slots: both tiles' global loads issue in one burst at
// the top of each iteration and are both stored to LDS before the S-WMMAs.
__global__ __launch_bounds__(256) void attn_fwd(
    const unsigned short* __restrict__ qbuf, const unsigned short* __restrict__ kbuf,
    const unsigned short* __restrict__ vbuf, float* __restrict__ out) {
  __shared__ unsigned short qs[WPB][16 * HEAD];   // 16 KB
  __shared__ unsigned short ks[WPB][32 * HEAD];   // 32 KB
  __shared__ unsigned short vs[WPB][32 * HEAD];   // 32 KB
  __shared__ unsigned short ps[WPB][16 * 32];     //  8 KB (P transpose staging)
  int lane = threadIdx.x & 31;
  int wid  = threadIdx.x >> 5;
  int tile = blockIdx.x * WPB + wid;              // 0..2047
  int b     = tile >> 7;
  int qBase = (tile & 127) * 16;

  unsigned short* qw = qs[wid];
  unsigned short* kw = ks[wid];
  unsigned short* vw = vs[wid];
  unsigned short* pw = ps[wid];

  const unsigned short* qg = qbuf + ((size_t)(b * TT + qBase)) * HEAD;
#pragma unroll
  for (int i = 0; i < 4; ++i)
    ((uint4*)qw)[i * 32 + lane] = ((const uint4*)qg)[i * 32 + lane];
  asm volatile("s_wait_dscnt 0" ::: "memory");

  v16bf qa0 = ldsAfrag(qw, HEAD, 0, lane);
  v16bf qa1 = ldsAfrag(qw, HEAD, 32, lane);

  v8f o[4];
#pragma unroll
  for (int nt = 0; nt < 4; ++nt) o[nt] = zero8();
  float mrow[8], lrow[8];
#pragma unroll
  for (int r = 0; r < 8; ++r) { mrow[r] = -INFINITY; lrow[r] = 0.0f; }

  int n   = lane & 15;
  int hi8 = (lane < 16) ? 0 : 8;
  int jmax = (qBase + 15) >> 5;                   // last 32-key tile (causal)

  for (int j = 0; j <= jmax; ++j) {
    int kBase = j * 32;
    const unsigned short* kg = kbuf + ((size_t)(b * TT + kBase)) * HEAD;
    const unsigned short* vg = vbuf + ((size_t)(b * TT + kBase)) * HEAD;

    // Stage K and V tiles (each 32x64 bf16) into their own LDS slots.
#pragma unroll
    for (int i = 0; i < 8; ++i)
      ((uint4*)kw)[i * 32 + lane] = ((const uint4*)kg)[i * 32 + lane];
#pragma unroll
    for (int i = 0; i < 8; ++i)
      ((uint4*)vw)[i * 32 + lane] = ((const uint4*)vg)[i * 32 + lane];
    if (j < jmax) {  // global_prefetch_b8 of next K/V tiles
      __builtin_prefetch(kg + (size_t)32 * HEAD, 0, 1);
      __builtin_prefetch(vg + (size_t)32 * HEAD, 0, 1);
    }
    asm volatile("s_wait_dscnt 0" ::: "memory");

    // S[16x32] = Q[16x64] x K^T[64x32], as two 16x16 f32 tiles.
    v16bf bk00 = ldsBTfrag(kw, HEAD, 0,  0,  lane);
    v16bf bk01 = ldsBTfrag(kw, HEAD, 0,  32, lane);
    v16bf bk10 = ldsBTfrag(kw, HEAD, 16, 0,  lane);
    v16bf bk11 = ldsBTfrag(kw, HEAD, 16, 32, lane);
    v8f s0 = zero8(), s1 = zero8();
    s0 = wmma_bf16(qa0, bk00, s0);
    s0 = wmma_bf16(qa1, bk01, s0);
    s1 = wmma_bf16(qa0, bk10, s1);
    s1 = wmma_bf16(qa1, bk11, s1);

    // Causal mask + online softmax (rows live in 16-lane halves).
#pragma unroll
    for (int r = 0; r < 8; ++r) {
      int m = r + hi8;
      int qrow = qBase + m;
      float x0 = (kBase + n      <= qrow) ? s0[r] : -INFINITY;
      float x1 = (kBase + 16 + n <= qrow) ? s1[r] : -INFINITY;
      float rmax = fmaxf(x0, x1);
#pragma unroll
      for (int msk = 1; msk < 16; msk <<= 1)
        rmax = fmaxf(rmax, __shfl_xor(rmax, msk, 32));
      float mold = mrow[r];
      float mnew = fmaxf(mold, rmax);     // rmax finite (diagonal always present)
      float sc = __expf(mold - mnew);
      mrow[r] = mnew;
      float p0 = __expf(x0 - mnew);
      float p1 = __expf(x1 - mnew);
      float rs = p0 + p1;
#pragma unroll
      for (int msk = 1; msk < 16; msk <<= 1)
        rs += __shfl_xor(rs, msk, 32);
      lrow[r] = lrow[r] * sc + rs;
#pragma unroll
      for (int nt = 0; nt < 4; ++nt) o[nt][r] *= sc;
      // Stage P (C-layout -> row-major) for the A-fragment transpose.
      pw[m * 32 + n]      = f2bf(p0);
      pw[m * 32 + 16 + n] = f2bf(p1);
    }
    asm volatile("s_wait_dscnt 0" ::: "memory");
    v16bf pa = ldsAfrag(pw, 32, 0, lane);

    // O[16x64] += P[16x32] x V[32x64]
#pragma unroll
    for (int nt = 0; nt < 4; ++nt) {
      v16bf bv = ldsBfrag(vw, HEAD, 0, nt * 16, lane);
      o[nt] = wmma_bf16(pa, bv, o[nt]);
    }
  }

  // Normalize and store fp32 output.
  float* og = out + ((size_t)(b * TT + qBase)) * HEAD;
#pragma unroll
  for (int nt = 0; nt < 4; ++nt) {
#pragma unroll
    for (int r = 0; r < 8; ++r) {
      int m = r + hi8;
      og[m * HEAD + nt * 16 + n] = o[nt][r] / lrow[r];
    }
  }
}

// ---- Host launcher ----------------------------------------------------------
// Inputs (setup_inputs order): x, Wk, bk, Wq, bq, Wv, bv  (all fp32)
// Workspace layout (u16 elements):
//   [0 .. 24576)           packed W fragments (48 KB)
//   [32768 .. +B*T*H)      Q (bf16, pre-scaled by 1/sqrt(H))
//   next B*T*H             K (bf16)
//   next B*T*H             V (bf16)
// Total ~12.1 MB of d_ws.
extern "C" void kernel_launch(void* const* d_in, const int* in_sizes, int n_in,
                              void* d_out, int out_size, void* d_ws, size_t ws_size,
                              hipStream_t stream) {
  const float* x  = (const float*)d_in[0];
  const float* Wk = (const float*)d_in[1];
  const float* bk = (const float*)d_in[2];
  const float* Wq = (const float*)d_in[3];
  const float* bq = (const float*)d_in[4];
  const float* Wv = (const float*)d_in[5];
  const float* bv = (const float*)d_in[6];
  float* out = (float*)d_out;

  unsigned short* wsu   = (unsigned short*)d_ws;
  unsigned short* wfrag = wsu;
  unsigned short* qb    = wsu + 32768;
  unsigned short* kbuf  = qb   + (size_t)BB * TT * HEAD;
  unsigned short* vbuf  = kbuf + (size_t)BB * TT * HEAD;

  prep_wfrag<<<48, 32, 0, stream>>>(Wq, Wk, Wv, wfrag);
  proj_qkv<<<(BB * TT / 16) / WPB, 32 * WPB, 0, stream>>>(x, bq, bk, bv, wfrag, qb, kbuf, vbuf);
  attn_fwd<<<(BB * TT / 16) / WPB, 32 * WPB, 0, stream>>>(qb, kbuf, vbuf, out);
}
